// NoShift_Doppler_Fourier_Net_18811956756733
// MI455X (gfx1250) — compile-verified, matrix operand order checked by reading, not run on hardware
//
#include <hip/hip_runtime.h>

// CDNA5 / gfx1250 wave32 WMMA types
typedef __attribute__((ext_vector_type(16))) _Float16 v16h;
typedef __attribute__((ext_vector_type(8)))  _Float16 v8h;
typedef __attribute__((ext_vector_type(4)))  _Float16 v4h;
typedef __attribute__((ext_vector_type(8)))  float    v8f;
typedef __attribute__((ext_vector_type(4)))  float    v4f;

#define KDIM        256          // DFT size (K and N of the GEMM)
#define LDS_STRIDE  264          // 256 + 8 halves padding -> 528B rows, conflict-free b128 reads
#define THREADS     512          // 16 waves: 4 (M super-tiles) x 4 (N-strips)
#define MTILES      2            // 2 x 16 = 32 rows per wave: each LDS B-frag feeds 2 WMMAs

extern __shared__ _Float16 lds_w[];  // [2][256][LDS_STRIDE] f16 = 264 KB (CDNA5 320KB LDS)

// Load 8+8 fp32 (two K-runs of a 16-bit A-tile row) and convert to one v16h.
// Layout per ISA "16-bit A-Matrix 16x32": lane<16 holds K = {0..7, 16..23},
// lane>=16 holds K = {8..15, 24..31}; caller bakes the +8 into the pointer.
__device__ inline v16h load_cvt_a(const float* __restrict__ p) {
  v4f a = *(const v4f*)(p);
  v4f b = *(const v4f*)(p + 4);
  v4f c = *(const v4f*)(p + 16);
  v4f d = *(const v4f*)(p + 20);
  v16h r;
  r[0]  = (_Float16)a.x;  r[1]  = (_Float16)a.y;  r[2]  = (_Float16)a.z;  r[3]  = (_Float16)a.w;
  r[4]  = (_Float16)b.x;  r[5]  = (_Float16)b.y;  r[6]  = (_Float16)b.z;  r[7]  = (_Float16)b.w;
  r[8]  = (_Float16)c.x;  r[9]  = (_Float16)c.y;  r[10] = (_Float16)c.z;  r[11] = (_Float16)c.w;
  r[12] = (_Float16)d.x;  r[13] = (_Float16)d.y;  r[14] = (_Float16)d.z;  r[15] = (_Float16)d.w;
  return r;
}

// Load a B-tile row fragment from LDS: two contiguous 16B runs (ds_load_b128 x2).
__device__ inline v16h load_b(const _Float16* __restrict__ p) {
  v8h lo = *(const v8h*)(p);
  v8h hi = *(const v8h*)(p + 16);
  return __builtin_shufflevector(lo, hi, 0,1,2,3,4,5,6,7,8,9,10,11,12,13,14,15);
}

__global__ __launch_bounds__(THREADS)
void dft_cplx_wmma(const float* __restrict__ xr, const float* __restrict__ xi,
                   const float* __restrict__ Wr, const float* __restrict__ Wi,
                   float* __restrict__ outR, float* __restrict__ outI) {
  _Float16* ldsWr = lds_w;
  _Float16* ldsWi = lds_w + KDIM * LDS_STRIDE;

  const int tid = threadIdx.x;

  // ---- Stage Wr/Wi into LDS as f16 (once per block; W stays L2-resident) ----
  for (int q = tid; q < (KDIM * KDIM) / 4; q += THREADS) {
    int n  = q >> 6;          // 64 float4 quads per 256-wide row
    int k4 = (q & 63) << 2;
    v4f r = *(const v4f*)(Wr + (size_t)n * KDIM + k4);
    v4f i = *(const v4f*)(Wi + (size_t)n * KDIM + k4);
    v4h hr, hi;
    hr.x = (_Float16)r.x; hr.y = (_Float16)r.y; hr.z = (_Float16)r.z; hr.w = (_Float16)r.w;
    hi.x = (_Float16)i.x; hi.y = (_Float16)i.y; hi.z = (_Float16)i.z; hi.w = (_Float16)i.w;
    *(v4h*)(ldsWr + n * LDS_STRIDE + k4) = hr;
    *(v4h*)(ldsWi + n * LDS_STRIDE + k4) = hi;
  }
  __syncthreads();

  // ---- Wave tiling: 16 waves = 4 M super-tiles (32 rows) x 4 N-strips (64 cols) ----
  const int wave    = tid >> 5;
  const int lane    = tid & 31;
  const int mSub    = wave >> 2;        // 0..3
  const int nStrip  = wave & 3;         // 0..3
  const int laneLow = lane & 15;
  const int kOff    = (lane >> 4) * 8;  // lane>=16 handles K-runs shifted by 8

  const size_t m0 = (size_t)blockIdx.x * (64 * MTILES) + (size_t)mSub * (16 * MTILES);
  const int    n0 = nStrip * 64;

  const float* xrRow = xr + (m0 + laneLow) * KDIM + kOff;
  const float* xiRow = xi + (m0 + laneLow) * KDIM + kOff;

  v8f accR[MTILES][4] = {}, accI[MTILES][4] = {};

  for (int k0 = 0; k0 < KDIM; k0 += 32) {
    // A-tiles for both 16-row m-tiles: xr, xi, and sign-flipped xi
    // (f16 WMMA NEG bits only cover C, so bake the minus into the operand)
    v16h Ar[MTILES], Ai[MTILES], Ani[MTILES];
#pragma unroll
    for (int mm = 0; mm < MTILES; ++mm) {
      const size_t moff = (size_t)mm * 16 * KDIM;
      Ar[mm]  = load_cvt_a(xrRow + moff + k0);
      Ai[mm]  = load_cvt_a(xiRow + moff + k0);
      Ani[mm] = -Ai[mm];   // packed sign-bit XOR
    }

#pragma unroll
    for (int n = 0; n < 4; ++n) {
      const int nrow = n0 + n * 16 + laneLow;
      const _Float16* pr = ldsWr + nrow * LDS_STRIDE + k0 + kOff;
      const _Float16* pi = ldsWi + nrow * LDS_STRIDE + k0 + kOff;
      v16h Br = load_b(pr);   // B[k][n] = W[n][k]  ->  computes x @ W^T
      v16h Bi = load_b(pi);

#pragma unroll
      for (int mm = 0; mm < MTILES; ++mm) {
        // out_r += xr*Wr^T + (-xi)*Wi^T ; out_i += xr*Wi^T + xi*Wr^T
        accR[mm][n] = __builtin_amdgcn_wmma_f32_16x16x32_f16(false, Ar[mm],  false, Br, (short)0, accR[mm][n], false, false);
        accR[mm][n] = __builtin_amdgcn_wmma_f32_16x16x32_f16(false, Ani[mm], false, Bi, (short)0, accR[mm][n], false, false);
        accI[mm][n] = __builtin_amdgcn_wmma_f32_16x16x32_f16(false, Ar[mm],  false, Bi, (short)0, accI[mm][n], false, false);
        accI[mm][n] = __builtin_amdgcn_wmma_f32_16x16x32_f16(false, Ai[mm],  false, Br, (short)0, accI[mm][n], false, false);
      }
    }
  }

  // ---- Store: C/D layout -> lanes 0-15 hold M=j, lanes 16-31 hold M=j+8, N=lane&15 ----
#pragma unroll
  for (int mm = 0; mm < MTILES; ++mm) {
    const size_t rowBase = m0 + (size_t)mm * 16 + (size_t)((lane >> 4) * 8);
#pragma unroll
    for (int n = 0; n < 4; ++n) {
      const int col = n0 + n * 16 + laneLow;
#pragma unroll
      for (int j = 0; j < 8; ++j) {
        const size_t off = (rowBase + j) * KDIM + col;
        outR[off] = accR[mm][n][j];
        outI[off] = accI[mm][n][j];
      }
    }
  }
}

extern "C" void kernel_launch(void* const* d_in, const int* in_sizes, int n_in,
                              void* d_out, int out_size, void* d_ws, size_t ws_size,
                              hipStream_t stream) {
  const float* xr = (const float*)d_in[0];   // [4,256,256,256] fp32
  const float* xi = (const float*)d_in[1];
  const float* Wr = (const float*)d_in[2];   // [256,256] fp32
  const float* Wi = (const float*)d_in[3];

  float* outR = (float*)d_out;               // out_r then out_i, flat
  float* outI = (float*)d_out + (size_t)in_sizes[0];

  const int M = in_sizes[0] / KDIM;          // 262,144 rows
  const int blocks = M / (64 * MTILES);      // 128 rows per workgroup -> 2048 blocks
  const size_t shmem = 2u * KDIM * LDS_STRIDE * 2u;  // 264 KB dynamic LDS

  dft_cplx_wmma<<<blocks, THREADS, shmem, stream>>>(xr, xi, Wr, Wi, outR, outI);
}
